// SelfAttention_old_13855564497213
// MI455X (gfx1250) — matrix operand order repaired and checked by reading, not compiled, not run on hardware
//
#include <hip/hip_runtime.h>
#include <math.h>

typedef __attribute__((ext_vector_type(16))) _Float16 v16h;
typedef __attribute__((ext_vector_type(8)))  float    v8f;
typedef _Float16 half_t;

#define BATCH 4
#define SEQ   4096
#define CH    512
#define CD    64     // bottleneck dim C/8
#define ROWS  (BATCH * SEQ)   // 16384
#define NCHUNK 8             // query-axis split for softmax stats
#define CHUNK  (SEQ / NCHUNK)

// ---------------------------------------------------------------------------
// WMMA fragment loaders (wave32, v_wmma_f32_16x16x32_f16 layouts per CDNA5 ISA)
// A (16x32, f16): lane l holds row M = l%16; h = l/16:
//   elements 0..7  -> K = h*8 + j
//   elements 8..15 -> K = 16 + h*8 + (j-8)
// B (32x16, f16): lane l holds col N = l%16; element j -> K = (l/16)*8 + j (j<8), 16 + (l/16)*8 + (j-8) (j>=8)
// C/D (16x16, f32): VGPR r, lane l -> M = r + 8*(l/16), N = l%16
// ---------------------------------------------------------------------------

__device__ __forceinline__ v16h load_A_f16(const half_t* src, int ld, int row0, int k0) {
    int l = threadIdx.x & 31;
    const half_t* p = src + (size_t)(row0 + (l & 15)) * ld + k0 + ((l >> 4) & 1) * 8;
    v16h a;
#pragma unroll
    for (int j = 0; j < 8; ++j) a[j] = p[j];
#pragma unroll
    for (int j = 0; j < 8; ++j) a[8 + j] = p[16 + j];
    return a;
}

__device__ __forceinline__ v16h load_A_f32cvt(const float* src, int ld, int row0, int k0) {
    int l = threadIdx.x & 31;
    const float* p = src + (size_t)(row0 + (l & 15)) * ld + k0 + ((l >> 4) & 1) * 8;
    v16h a;
#pragma unroll
    for (int j = 0; j < 8; ++j) a[j] = (half_t)p[j];
#pragma unroll
    for (int j = 0; j < 8; ++j) a[8 + j] = (half_t)p[16 + j];
    return a;
}

// B tile where logical B = src^T:  B[k][n] = src[n][k]  (src row-major NxK f16)
__device__ __forceinline__ v16h load_B_T_f16(const half_t* src, int ld, int col0, int k0) {
    int l = threadIdx.x & 31;
    const half_t* p = src + (size_t)(col0 + (l & 15)) * ld + k0 + ((l >> 4) & 1) * 8;
    v16h b;
#pragma unroll
    for (int j = 0; j < 16; ++j) b[j] = p[j < 8 ? j : j + 8];
    return b;
}

// B tile from row-major KxN f32 weight (convert to f16).
__device__ __forceinline__ v16h load_B_rm_f32(const float* src, int ld, int k0, int col0) {
    int l = threadIdx.x & 31;
    const float* p = src + (size_t)(k0 + ((l >> 4) & 1) * 8) * ld + col0 + (l & 15);
    v16h b;
#pragma unroll
    for (int j = 0; j < 16; ++j) b[j] = (half_t)p[(size_t)(j < 8 ? j : j + 8) * ld];
    return b;
}

// B tile from row-major KxN f16 source.
__device__ __forceinline__ v16h load_B_rm_f16(const half_t* src, int ld, int k0, int col0) {
    int l = threadIdx.x & 31;
    const half_t* p = src + (size_t)(k0 + ((l >> 4) & 1) * 8) * ld + col0 + (l & 15);
    v16h b;
#pragma unroll
    for (int j = 0; j < 16; ++j) b[j] = p[(size_t)(j < 8 ? j : j + 8) * ld];
    return b;
}

__device__ __forceinline__ void store_D_f16(half_t* dst, int ld, int row0, int col0, v8f d) {
    int l = threadIdx.x & 31;
    int col = col0 + (l & 15);
    int rb  = row0 + ((l >> 4) & 1) * 8;
#pragma unroll
    for (int r = 0; r < 8; ++r) dst[(size_t)(rb + r) * ld + col] = (half_t)d[r];
}

__device__ __forceinline__ void store_D_f32(float* dst, int ld, int row0, int col0, v8f d) {
    int l = threadIdx.x & 31;
    int col = col0 + (l & 15);
    int rb  = row0 + ((l >> 4) & 1) * 8;
#pragma unroll
    for (int r = 0; r < 8; ++r) dst[(size_t)(rb + r) * ld + col] = d[r];
}

__device__ __forceinline__ v8f wmma_f16(v16h a, v16h b, v8f c) {
    return __builtin_amdgcn_wmma_f32_16x16x32_f16(false, a, false, b, (short)0, c, false, false);
}

// ---------------------------------------------------------------------------
// Pass 1: q/k/v = x @ W{q,k,v}   (ROWSx512)*(512x64) -> f16 in workspace
// grid (ROWS/16, 3), block 128: wave w = column tile w (4 x 16 = 64 cols)
// ---------------------------------------------------------------------------
__global__ void qkv_kernel(const float* __restrict__ x,
                           const float* __restrict__ Wq,
                           const float* __restrict__ Wk,
                           const float* __restrict__ Wv,
                           half_t* __restrict__ qh,
                           half_t* __restrict__ kh,
                           half_t* __restrict__ vh) {
    const int wave = threadIdx.x >> 5;
    const int row0 = blockIdx.x * 16;
    const int col0 = wave * 16;
    const float* W;
    half_t* dst;
    switch (blockIdx.y) {
        case 0: W = Wq; dst = qh; break;
        case 1: W = Wk; dst = kh; break;
        default: W = Wv; dst = vh; break;
    }
    v8f acc = {};
#pragma unroll 4
    for (int kk = 0; kk < CH; kk += 32) {
        v16h a = load_A_f32cvt(x, CH, row0, kk);
        v16h b = load_B_rm_f32(W, CD, kk, col0);
        acc = wmma_f16(a, b, acc);
    }
    store_D_f16(dst, CD, row0, col0, acc);
}

// ---------------------------------------------------------------------------
// Pass 2: logits[b,n,m] = q[b,n,:] . k[b,m,:]   (written raw into d_out attn)
// grid (SEQ/16/8, SEQ/16, BATCH), block 256: wave = one 16x16 (n,m) tile
// ---------------------------------------------------------------------------
__global__ void logits_kernel(const half_t* __restrict__ qh,
                              const half_t* __restrict__ kh,
                              float* __restrict__ logits) {
    const int wave = threadIdx.x >> 5;
    const int n0 = blockIdx.y * 16;
    const int m0 = (blockIdx.x * 8 + wave) * 16;
    const int b  = blockIdx.z;
    const half_t* qb = qh + (size_t)b * SEQ * CD;
    const half_t* kb = kh + (size_t)b * SEQ * CD;
    float* lg = logits + (size_t)b * SEQ * SEQ;
    v8f acc = {};
#pragma unroll
    for (int kk = 0; kk < CD; kk += 32) {
        v16h a = load_A_f16(qb, CD, n0, kk);
        v16h bm = load_B_T_f16(kb, CD, m0, kk);
        acc = wmma_f16(a, bm, acc);
    }
    store_D_f32(lg, SEQ, n0, m0, acc);
}

// ---------------------------------------------------------------------------
// Pass 3a: partial per-column softmax stats over a CHUNK of the query axis.
// grid (SEQ/256, NCHUNK, BATCH), block 256: one thread per column, 8 chunks
// in parallel -> 4096 waves of MLP for a 268MB streaming pass.
// ---------------------------------------------------------------------------
__global__ void colstats_partial_kernel(const float* __restrict__ logits,
                                        float* __restrict__ pmax,
                                        float* __restrict__ psum) {
    const int b = blockIdx.z;
    const int c = blockIdx.y;
    const int m = blockIdx.x * 256 + threadIdx.x;
    const float* p = logits + (size_t)b * SEQ * SEQ + (size_t)c * CHUNK * SEQ + m;
    float mx = -INFINITY, s = 0.f;
#pragma unroll 4
    for (int n = 0; n < CHUNK; ++n) {
        int np = (n + 16 < CHUNK) ? n + 16 : CHUNK - 1;
        __builtin_prefetch(p + (size_t)np * SEQ, 0, 1);
        float v = p[(size_t)n * SEQ];
        float mn = fmaxf(mx, v);
        s = s * __expf(mx - mn) + __expf(v - mn);
        mx = mn;
    }
    const size_t o = ((size_t)b * NCHUNK + c) * SEQ + m;
    pmax[o] = mx;
    psum[o] = s;
}

// ---------------------------------------------------------------------------
// Pass 3b: combine the NCHUNK partial stats per column.
// ---------------------------------------------------------------------------
__global__ void colstats_combine_kernel(const float* __restrict__ pmax,
                                        const float* __restrict__ psum,
                                        float* __restrict__ colmax,
                                        float* __restrict__ colrinv) {
    const int b = blockIdx.y;
    const int m = blockIdx.x * 256 + threadIdx.x;
    float mx = -INFINITY, s = 0.f;
#pragma unroll
    for (int c = 0; c < NCHUNK; ++c) {
        const size_t o = ((size_t)b * NCHUNK + c) * SEQ + m;
        float pm = pmax[o];
        float ps = psum[o];
        float mn = fmaxf(mx, pm);
        s = s * __expf(mx - mn) + ps * __expf(pm - mn);
        mx = mn;
    }
    colmax[b * SEQ + m]  = mx;
    colrinv[b * SEQ + m] = 1.f / s;
}

// ---------------------------------------------------------------------------
// Pass 4 (fused): afm = attn^T @ v with softmax normalization fused in.
// Reads RAW logits (each element exactly once), computes
//   p = exp(l - colmax[m]) * colrinv[m],
// writes p back in place (final attn output) and feeds p into the WMMA
// A-fragment. K (query) axis split 4-ways across the block's waves for 4x
// memory-level parallelism; partial accumulators reduced through LDS.
// grid (SEQ/16, BATCH), block 128: wave w owns K-range [w*1024, w*1024+1024).
// ---------------------------------------------------------------------------
__global__ void afm_fused_kernel(float* __restrict__ attn,   // in: logits, out: attn
                                 const half_t* __restrict__ vh,
                                 const float* __restrict__ colmax,
                                 const float* __restrict__ colrinv,
                                 float* __restrict__ afm) {
    const int wave = threadIdx.x >> 5;
    const int b  = blockIdx.y;
    const int m0 = blockIdx.x * 16;
    float* ab = attn + (size_t)b * SEQ * SEQ;
    const half_t* vb = vh + (size_t)b * SEQ * CD;
    float* db = afm + (size_t)b * SEQ * CD;

    const int l = threadIdx.x & 31;
    const int m = m0 + (l & 15);          // fixed column per lane
    const int h = (l >> 4) & 1;
    const float mx = colmax[b * SEQ + m];
    const float ri = colrinv[b * SEQ + m];

    v8f acc[4] = {{}, {}, {}, {}};
    const int kbeg = wave * (SEQ / 4);
    const int kend = kbeg + (SEQ / 4);
    for (int k0 = kbeg; k0 < kend; k0 += 32) {
        // gather 16 raw logits for this lane's (m, k-range), normalize,
        // write attn back in place, pack into the A fragment
        v16h a;
#pragma unroll
        for (int j = 0; j < 8; ++j) {
            size_t o = (size_t)(k0 + h * 8 + j) * SEQ + m;
            float pv = __expf(ab[o] - mx) * ri;
            ab[o] = pv;
            a[j] = (half_t)pv;
        }
#pragma unroll
        for (int j = 0; j < 8; ++j) {
            size_t o = (size_t)(k0 + 16 + h * 8 + j) * SEQ + m;
            float pv = __expf(ab[o] - mx) * ri;
            ab[o] = pv;
            a[8 + j] = (half_t)pv;
        }
#pragma unroll
        for (int t = 0; t < 4; ++t) {
            v16h bf = load_B_rm_f16(vb, CD, k0, t * 16);
            acc[t] = wmma_f16(a, bf, acc[t]);
        }
    }

    // block reduction of the 4 waves' partial tiles through LDS
    __shared__ float red[4][4][256];      // [wave][tile][r*32 + lane]
#pragma unroll
    for (int t = 0; t < 4; ++t)
#pragma unroll
        for (int r = 0; r < 8; ++r)
            red[wave][t][r * 32 + l] = acc[t][r];
    __syncthreads();

    // wave w finalizes output tile w
    v8f s = {};
#pragma unroll
    for (int r = 0; r < 8; ++r)
        s[r] = red[0][wave][r * 32 + l] + red[1][wave][r * 32 + l] +
               red[2][wave][r * 32 + l] + red[3][wave][r * 32 + l];
    store_D_f32(db, CD, m0, wave * 16, s);
}

// ---------------------------------------------------------------------------
// Pass 5: out = x + (afm @ Wout) * gamma    (ROWSx64)*(64x512)
// grid (ROWS/16, 4), block 256: wave = one 16x16 tile of 512 cols
// ---------------------------------------------------------------------------
__global__ void outproj_kernel(const float* __restrict__ x,
                               const float* __restrict__ afm,
                               const float* __restrict__ Wout,
                               const float* __restrict__ gamma,
                               float* __restrict__ out) {
    const int wave = threadIdx.x >> 5;
    const int row0 = blockIdx.x * 16;
    const int col0 = (blockIdx.y * 8 + wave) * 16;
    v8f acc = {};
#pragma unroll
    for (int kk = 0; kk < CD; kk += 32) {
        v16h a = load_A_f32cvt(afm, CD, row0, kk);
        v16h b = load_B_rm_f32(Wout, CH, kk, col0);
        acc = wmma_f16(a, b, acc);
    }
    const float g = gamma[0];
    const int l = threadIdx.x & 31;
    const int col = col0 + (l & 15);
    const int rb  = row0 + ((l >> 4) & 1) * 8;
#pragma unroll
    for (int r = 0; r < 8; ++r) {
        size_t o = (size_t)(rb + r) * CH + col;
        out[o] = x[o] + g * acc[r];
    }
}

// ---------------------------------------------------------------------------
extern "C" void kernel_launch(void* const* d_in, const int* in_sizes, int n_in,
                              void* d_out, int out_size, void* d_ws, size_t ws_size,
                              hipStream_t stream) {
    const float* x     = (const float*)d_in[0];
    const float* Wq    = (const float*)d_in[1];
    const float* Wk    = (const float*)d_in[2];
    const float* Wv    = (const float*)d_in[3];
    const float* Wout  = (const float*)d_in[4];
    const float* gamma = (const float*)d_in[5];

    float* out  = (float*)d_out;
    float* attn = out + (size_t)BATCH * SEQ * CH;   // 8,388,608 floats in

    // workspace layout (~11.4 MB)
    char* w = (char*)d_ws;
    half_t* qh  = (half_t*)w;                    w += (size_t)ROWS * CD * sizeof(half_t);
    half_t* kh  = (half_t*)w;                    w += (size_t)ROWS * CD * sizeof(half_t);
    half_t* vh  = (half_t*)w;                    w += (size_t)ROWS * CD * sizeof(half_t);
    float*  afm = (float*)w;                     w += (size_t)ROWS * CD * sizeof(float);
    float*  colmax  = (float*)w;                 w += (size_t)BATCH * SEQ * sizeof(float);
    float*  colrinv = (float*)w;                 w += (size_t)BATCH * SEQ * sizeof(float);
    float*  pmax = (float*)w;                    w += (size_t)BATCH * NCHUNK * SEQ * sizeof(float);
    float*  psum = (float*)w;

    // 1) q/k/v projections (WMMA f16)
    qkv_kernel<<<dim3(ROWS / 16, 3), 128, 0, stream>>>(x, Wq, Wk, Wv, qh, kh, vh);

    // 2) logits = q @ k^T, written raw into the attn output region
    logits_kernel<<<dim3(SEQ / 16 / 8, SEQ / 16, BATCH), 256, 0, stream>>>(qh, kh, attn);

    // 3) softmax stats over query axis (per key column), chunked for MLP
    colstats_partial_kernel<<<dim3(SEQ / 256, NCHUNK, BATCH), 256, 0, stream>>>(attn, pmax, psum);
    colstats_combine_kernel<<<dim3(SEQ / 256, BATCH), 256, 0, stream>>>(pmax, psum, colmax, colrinv);

    // 4) fused normalize + afm = attn^T @ v (WMMA); finalizes attn in place
    afm_fused_kernel<<<dim3(SEQ / 16, BATCH), 128, 0, stream>>>(attn, vh, colmax, colrinv, afm);

    // 5) out = x + gamma * (afm @ Wout) (WMMA)
    outproj_kernel<<<dim3(ROWS / 16, 4), 256, 0, stream>>>(x, afm, Wout, gamma, out);
}
